// Attention_51410758533700
// MI455X (gfx1250) — compile-verified, matrix operand order checked by reading, not run on hardware
//
#include <hip/hip_runtime.h>
#include <hip/hip_bf16.h>
#include <math.h>

typedef __attribute__((ext_vector_type(16))) __bf16 v16bf;
typedef __attribute__((ext_vector_type(8)))  float  v8f;

// 128-bit chunk pointers matching __builtin_amdgcn_global_load_async_to_lds_b128:
//   param0: v4i in addrspace(1) (global), param1: v4i in addrspace(3) (LDS)
typedef int v4i_vec __attribute__((vector_size(16)));
typedef __attribute__((address_space(1))) v4i_vec* gv4p;
typedef __attribute__((address_space(3))) v4i_vec* lv4p;

#ifndef __has_builtin
#define __has_builtin(x) 0
#endif
#if __has_builtin(__builtin_amdgcn_global_load_async_to_lds_b128)
#define HAVE_ASYNC_LDS 1
#else
#define HAVE_ASYNC_LDS 0
#endif
#if __has_builtin(__builtin_amdgcn_s_wait_asynccnt)
#define WAIT_ASYNC() __builtin_amdgcn_s_wait_asynccnt(0)
#else
#define WAIT_ASYNC() asm volatile("s_wait_asynccnt 0" ::: "memory")
#endif
#if __has_builtin(__builtin_amdgcn_permlane16)
#define HAVE_PERMLANE16 1
#else
#define HAVE_PERMLANE16 0
#endif

#define B_  2
#define H_  16
#define N_  2048
#define D_  64
#define BM  64
#define BN  64

// ---- VALU cross-lane reductions within each 16-lane row (no LDS traffic) ----
#if HAVE_PERMLANE16
__device__ __forceinline__ float plane16(float x, unsigned s0, unsigned s1) {
  union { float f; unsigned u; } a, r;
  a.f = x;
  r.u = __builtin_amdgcn_permlane16(a.u, a.u, s0, s1, false, false);
  return r.f;
}
__device__ __forceinline__ float red16_max(float v) {
  v = fmaxf(v, plane16(v, 0x67452301u, 0xEFCDAB89u));  // xor 1
  v = fmaxf(v, plane16(v, 0x54761032u, 0xDCFE98BAu));  // xor 2
  v = fmaxf(v, plane16(v, 0x32107654u, 0xBA98FEDCu));  // xor 4
  v = fmaxf(v, plane16(v, 0xFEDCBA98u, 0x76543210u));  // xor 8
  return v;
}
__device__ __forceinline__ float red16_add(float v) {
  v += plane16(v, 0x67452301u, 0xEFCDAB89u);
  v += plane16(v, 0x54761032u, 0xDCFE98BAu);
  v += plane16(v, 0x32107654u, 0xBA98FEDCu);
  v += plane16(v, 0xFEDCBA98u, 0x76543210u);
  return v;
}
#else
__device__ __forceinline__ float red16_max(float v) {
  #pragma unroll
  for (int msk = 1; msk < 16; msk <<= 1) v = fmaxf(v, __shfl_xor(v, msk, 32));
  return v;
}
__device__ __forceinline__ float red16_add(float v) {
  #pragma unroll
  for (int msk = 1; msk < 16; msk <<= 1) v += __shfl_xor(v, msk, 32);
  return v;
}
#endif

// ---- 16x32 bf16 WMMA fragment from a row pointer ----
// A-matrix: lane&15 = M (caller selects row ptr); K pattern per ISA 7.12.2:
//   lane<16: K = col0 + {0..7, 16..23}; lane>=16: K = col0 + {8..15, 24..31}
// B-matrix mirrors with lane&15 = N.
__device__ __forceinline__ v16bf load_frag(const __bf16* rowp, int col0, int lane) {
  int b0 = (lane & 16) ? 8 : 0;
  const __bf16* p = rowp + col0 + b0;
  union { v16bf v; uint4 u[2]; } f;
  f.u[0] = *(const uint4*)p;
  f.u[1] = *(const uint4*)(p + 16);
  return f.v;
}

// ---- one wave: l2-normalize row of 64, per-dim scale, RoPE, *omul, bf16 store ----
__device__ __forceinline__ void prep_row(const float* __restrict__ row,
                                         const float* __restrict__ scale,
                                         float pos, float omul, int lane, __bf16* dst) {
  float2 x = ((const float2*)row)[lane];            // elements 2*lane, 2*lane+1
  float ss = x.x * x.x + x.y * x.y;
  #pragma unroll
  for (int m = 1; m < 32; m <<= 1) ss += __shfl_xor(ss, m, 32);
  float inv = 1.0f / fmaxf(sqrtf(ss), 1e-12f);
  int e0 = lane * 2;
  float y0 = x.x * inv * scale[e0];
  float y1 = x.y * inv * scale[e0 + 1];
  float p0 = __shfl_xor(y0, 16, 32);                // rotate_half partner (+/-32 elems)
  float p1 = __shfl_xor(y1, 16, 32);
  float r0 = (lane < 16) ? -p0 : p0;
  float r1 = (lane < 16) ? -p1 : p1;
  const float kLn = -9.210340371976184f / 32.0f;    // -ln(10000)/32
  int j0 = e0 & 31;
  float f0 = pos * __expf((float)j0 * kLn);
  float f1 = pos * __expf((float)(j0 + 1) * kLn);
  union { __bf16 h[2]; unsigned u; } pk;
  pk.h[0] = (__bf16)((y0 * __cosf(f0) + r0 * __sinf(f0)) * omul);
  pk.h[1] = (__bf16)((y1 * __cosf(f1) + r1 * __sinf(f1)) * omul);
  ((unsigned*)dst)[lane] = pk.u;
}

// ================= pass 1a: Q/K norm + rope -> bf16 (Q pre-scaled by d^-1/2) ======
__global__ __launch_bounds__(128) void prep_qk_kernel(
    const float* __restrict__ q, const float* __restrict__ k,
    const float* __restrict__ qs, const float* __restrict__ ks,
    __bf16* __restrict__ qb, __bf16* __restrict__ kb) {
  const int lane = threadIdx.x & 31;
  const size_t ri = (size_t)blockIdx.x * 4 + (threadIdx.x >> 5); // row over B*H*N
  const float pos = (float)(ri % N_);
  prep_row(q + ri * D_, qs, pos, 0.125f, lane, qb + ri * D_);
  prep_row(k + ri * D_, ks, pos, 1.0f,  lane, kb + ri * D_);
}

// ================= pass 1b: V -> bf16, transposed [bh][d][n] =================
__global__ __launch_bounds__(128) void prep_v_kernel(const float* __restrict__ v,
                                                     __bf16* __restrict__ vt) {
  __shared__ __align__(16) __bf16 t[D_][BN];
  const int bh = blockIdx.y;
  const int n0 = blockIdx.x * BN;
  const float* vb = v + ((size_t)bh * N_ + n0) * D_;
  for (int i = threadIdx.x; i < BN * D_; i += 128) {
    int key = i >> 6, dd = i & 63;
    t[dd][key] = (__bf16)vb[(size_t)key * D_ + dd];
  }
  __syncthreads();
  __bf16* o = vt + (size_t)bh * D_ * N_;
  for (int i = threadIdx.x; i < BN * D_; i += 128) {
    int dd = i >> 6, c = i & 63;
    o[(size_t)dd * N_ + n0 + c] = t[dd][c];
  }
}

// ================= pass 2: flash attention on prepped bf16 =================
__global__ __launch_bounds__(128) void flash_bf16_cdna5(
    const __bf16* __restrict__ qb, const __bf16* __restrict__ kb,
    const __bf16* __restrict__ vt, float* __restrict__ out) {
  __shared__ __align__(16) __bf16 ktile[2][BN][D_];   // [key][d]  (B for S)
  __shared__ __align__(16) __bf16 vtile[2][D_][BN];   // [d][key]  (B for PV)
  __shared__ __align__(16) __bf16 ptile[4][16][BN];   // per-wave P scratch

  const int lane = threadIdx.x & 31;
  const int wave = threadIdx.x >> 5;
  const int q0 = blockIdx.x * BM;
  const int h  = blockIdx.y;
  const int b  = blockIdx.z;
  const int bh = b * H_ + h;

  const __bf16* kbase = kb + (size_t)bh * N_ * D_;
  const __bf16* vbase = vt + (size_t)bh * D_ * N_;    // [d][n]

  // Q A-fragments straight from global (rows are contiguous 128B)
  const __bf16* qrow = qb + ((size_t)bh * N_ + q0 + wave * 16 + (lane & 15)) * D_;
  const v16bf qa0 = load_frag(qrow, 0,  lane);
  const v16bf qa1 = load_frag(qrow, 32, lane);

  // double-buffered async stage of K tile (contiguous 8KB) and V^T tile (64x128B rows)
  auto stage = [&](int buf, int kb0) {
#if HAVE_ASYNC_LDS
    gv4p gk = (gv4p)(void*)(kbase + (size_t)kb0 * D_);
    lv4p lk = (lv4p)(void*)&ktile[buf][0][0];
    #pragma unroll
    for (int c0 = 0; c0 < 512; c0 += 128) {
      int c = c0 + (int)threadIdx.x;
      __builtin_amdgcn_global_load_async_to_lds_b128(gk + c, lk + c, 0, 0);
    }
    lv4p lv = (lv4p)(void*)&vtile[buf][0][0];
    #pragma unroll
    for (int c0 = 0; c0 < 512; c0 += 128) {
      int c = c0 + (int)threadIdx.x;
      int dd = c >> 3, c16 = c & 7;
      gv4p gvp = (gv4p)(void*)(vbase + (size_t)dd * N_ + kb0 + c16 * 8);
      __builtin_amdgcn_global_load_async_to_lds_b128(gvp, lv + c, 0, 0);
    }
#else
    const uint4* gk = (const uint4*)(kbase + (size_t)kb0 * D_);
    for (int c = threadIdx.x; c < 512; c += 128) {
      ((uint4*)&ktile[buf][0][0])[c] = gk[c];
      int dd = c >> 3, c16 = c & 7;
      ((uint4*)&vtile[buf][0][0])[c] =
          *(const uint4*)(vbase + (size_t)dd * N_ + kb0 + c16 * 8);
    }
#endif
  };

  float mrow[8], lrow[8];
  v8f o[4];
  #pragma unroll
  for (int r = 0; r < 8; ++r) { mrow[r] = -__builtin_inff(); lrow[r] = 0.f; }
  #pragma unroll
  for (int nb = 0; nb < 4; ++nb)
    #pragma unroll
    for (int r = 0; r < 8; ++r) o[nb][r] = 0.f;

  __bf16* pmine = &ptile[wave][0][0];
  const int hof = (lane & 16) ? 8 : 0;

  stage(0, 0);
  for (int it = 0; it < N_ / BN; ++it) {
    const int buf = it & 1;
    WAIT_ASYNC();            // this wave's async chunks for `buf` complete
    __syncthreads();         // all waves' chunks visible; prev compute finished
    if (it + 1 < N_ / BN) stage(buf ^ 1, (it + 1) * BN);  // prefetch next tile

    // ---- S = Q K^T  (Q carries the d^-1/2 scale) ----
    v8f s[4];
    #pragma unroll
    for (int nb = 0; nb < 4; ++nb) {
      v8f acc;
      #pragma unroll
      for (int r = 0; r < 8; ++r) acc[r] = 0.f;
      v16bf kf = load_frag(&ktile[buf][nb * 16 + (lane & 15)][0], 0, lane);
      acc = __builtin_amdgcn_wmma_f32_16x16x32_bf16(false, qa0, false, kf, (short)0, acc, false, false);
      kf = load_frag(&ktile[buf][nb * 16 + (lane & 15)][0], 32, lane);
      acc = __builtin_amdgcn_wmma_f32_16x16x32_bf16(false, qa1, false, kf, (short)0, acc, false, false);
      s[nb] = acc;
    }

    // ---- online softmax (row = r + hof; VALU permlane reductions) ----
    float mnew[8], alpha[8], rs[8];
    #pragma unroll
    for (int r = 0; r < 8; ++r) {
      float mv = fmaxf(fmaxf(s[0][r], s[1][r]), fmaxf(s[2][r], s[3][r]));
      mv = red16_max(mv);
      mnew[r]  = fmaxf(mrow[r], mv);
      alpha[r] = __expf(mrow[r] - mnew[r]);
      mrow[r]  = mnew[r];
      rs[r]    = 0.f;
    }
    #pragma unroll
    for (int nb = 0; nb < 4; ++nb)
      #pragma unroll
      for (int r = 0; r < 8; ++r) {
        float p = __expf(s[nb][r] - mnew[r]);
        rs[r] += p;
        s[nb][r] = p;
      }
    #pragma unroll
    for (int r = 0; r < 8; ++r)
      lrow[r] = lrow[r] * alpha[r] + red16_add(rs[r]);

    // ---- P -> LDS (A-layout source), rescale O ----
    #pragma unroll
    for (int nb = 0; nb < 4; ++nb)
      #pragma unroll
      for (int r = 0; r < 8; ++r) {
        pmine[(r + hof) * BN + nb * 16 + (lane & 15)] = (__bf16)s[nb][r];
        o[nb][r] *= alpha[r];
      }

    // ---- O += P V ----
    v16bf pa0 = load_frag(pmine + (lane & 15) * BN, 0,  lane);
    v16bf pa1 = load_frag(pmine + (lane & 15) * BN, 32, lane);
    #pragma unroll
    for (int nb = 0; nb < 4; ++nb) {
      v16bf vf = load_frag(&vtile[buf][nb * 16 + (lane & 15)][0], 0, lane);
      o[nb] = __builtin_amdgcn_wmma_f32_16x16x32_bf16(false, pa0, false, vf, (short)0, o[nb], false, false);
      vf = load_frag(&vtile[buf][nb * 16 + (lane & 15)][0], 32, lane);
      o[nb] = __builtin_amdgcn_wmma_f32_16x16x32_bf16(false, pa1, false, vf, (short)0, o[nb], false, false);
    }
  }

  // ---- epilogue: O /= l, write b n (h d) ----
  #pragma unroll
  for (int r = 0; r < 8; ++r) {
    float invl = 1.0f / lrow[r];
    int i = q0 + wave * 16 + r + hof;
    float* orow = out + ((size_t)b * N_ + i) * (H_ * D_) + h * D_;
    #pragma unroll
    for (int nb = 0; nb < 4; ++nb)
      orow[nb * 16 + (lane & 15)] = o[nb][r] * invl;
  }
}

// ================= fallback: fully fused (used only if ws too small) =================
__global__ __launch_bounds__(128) void flash_fused_cdna5(
    const float* __restrict__ q, const float* __restrict__ k,
    const float* __restrict__ v, const float* __restrict__ qs,
    const float* __restrict__ ks, float* __restrict__ out) {
  __shared__ __align__(16) __bf16 qtile[BM][D_];
  __shared__ __align__(16) __bf16 ktile[BN][D_];
  __shared__ __align__(16) __bf16 vtile[D_][BN];
  __shared__ __align__(16) __bf16 ptile[4][16][BN];

  const int lane = threadIdx.x & 31;
  const int wave = threadIdx.x >> 5;
  const int q0 = blockIdx.x * BM;
  const int h  = blockIdx.y;
  const int b  = blockIdx.z;
  const size_t bh_off = (size_t)(b * H_ + h) * N_ * D_;
  const float* qbase = q + bh_off;
  const float* kbase = k + bh_off;
  const float* vbase = v + bh_off;

  #pragma unroll
  for (int r = 0; r < 16; ++r) {
    int row = wave * 16 + r;
    prep_row(qbase + (size_t)(q0 + row) * D_, qs, (float)(q0 + row), 0.125f, lane, &qtile[row][0]);
  }
  __syncthreads();
  const v16bf qa0 = load_frag(&qtile[wave * 16 + (lane & 15)][0], 0,  lane);
  const v16bf qa1 = load_frag(&qtile[wave * 16 + (lane & 15)][0], 32, lane);

  float mrow[8], lrow[8];
  v8f o[4];
  #pragma unroll
  for (int r = 0; r < 8; ++r) { mrow[r] = -__builtin_inff(); lrow[r] = 0.f; }
  #pragma unroll
  for (int nb = 0; nb < 4; ++nb)
    #pragma unroll
    for (int r = 0; r < 8; ++r) o[nb][r] = 0.f;

  __bf16* pmine = &ptile[wave][0][0];
  const int hof = (lane & 16) ? 8 : 0;

  for (int kb0 = 0; kb0 < N_; kb0 += BN) {
    #pragma unroll
    for (int r = 0; r < 16; ++r) {
      int row = wave * 16 + r;
      prep_row(kbase + (size_t)(kb0 + row) * D_, ks, (float)(kb0 + row), 1.0f, lane, &ktile[row][0]);
    }
    for (int i = threadIdx.x; i < BN * D_; i += 128) {
      int key = i >> 6, dd = i & 63;
      vtile[dd][key] = (__bf16)vbase[(size_t)(kb0 + key) * D_ + dd];
    }
    __syncthreads();

    v8f s[4];
    #pragma unroll
    for (int nb = 0; nb < 4; ++nb) {
      v8f acc;
      #pragma unroll
      for (int r = 0; r < 8; ++r) acc[r] = 0.f;
      v16bf kf = load_frag(&ktile[nb * 16 + (lane & 15)][0], 0, lane);
      acc = __builtin_amdgcn_wmma_f32_16x16x32_bf16(false, qa0, false, kf, (short)0, acc, false, false);
      kf = load_frag(&ktile[nb * 16 + (lane & 15)][0], 32, lane);
      acc = __builtin_amdgcn_wmma_f32_16x16x32_bf16(false, qa1, false, kf, (short)0, acc, false, false);
      s[nb] = acc;
    }

    float mnew[8], alpha[8], rs[8];
    #pragma unroll
    for (int r = 0; r < 8; ++r) {
      float mv = fmaxf(fmaxf(s[0][r], s[1][r]), fmaxf(s[2][r], s[3][r]));
      mv = red16_max(mv);
      mnew[r]  = fmaxf(mrow[r], mv);
      alpha[r] = __expf(mrow[r] - mnew[r]);
      mrow[r]  = mnew[r];
      rs[r]    = 0.f;
    }
    #pragma unroll
    for (int nb = 0; nb < 4; ++nb)
      #pragma unroll
      for (int r = 0; r < 8; ++r) {
        float p = __expf(s[nb][r] - mnew[r]);
        rs[r] += p;
        s[nb][r] = p;
      }
    #pragma unroll
    for (int r = 0; r < 8; ++r)
      lrow[r] = lrow[r] * alpha[r] + red16_add(rs[r]);
    #pragma unroll
    for (int nb = 0; nb < 4; ++nb)
      #pragma unroll
      for (int r = 0; r < 8; ++r) {
        pmine[(r + hof) * BN + nb * 16 + (lane & 15)] = (__bf16)s[nb][r];
        o[nb][r] *= alpha[r];
      }

    v16bf pa0 = load_frag(pmine + (lane & 15) * BN, 0,  lane);
    v16bf pa1 = load_frag(pmine + (lane & 15) * BN, 32, lane);
    #pragma unroll
    for (int nb = 0; nb < 4; ++nb) {
      v16bf vf = load_frag(&vtile[nb * 16 + (lane & 15)][0], 0, lane);
      o[nb] = __builtin_amdgcn_wmma_f32_16x16x32_bf16(false, pa0, false, vf, (short)0, o[nb], false, false);
      vf = load_frag(&vtile[nb * 16 + (lane & 15)][0], 32, lane);
      o[nb] = __builtin_amdgcn_wmma_f32_16x16x32_bf16(false, pa1, false, vf, (short)0, o[nb], false, false);
    }
    __syncthreads();
  }

  #pragma unroll
  for (int r = 0; r < 8; ++r) {
    float invl = 1.0f / lrow[r];
    int i = q0 + wave * 16 + r + hof;
    float* orow = out + ((size_t)b * N_ + i) * (H_ * D_) + h * D_;
    #pragma unroll
    for (int nb = 0; nb < 4; ++nb)
      orow[nb * 16 + (lane & 15)] = o[nb][r] * invl;
  }
}

extern "C" void kernel_launch(void* const* d_in, const int* in_sizes, int n_in,
                              void* d_out, int out_size, void* d_ws, size_t ws_size,
                              hipStream_t stream) {
  (void)in_sizes; (void)n_in; (void)out_size;
  const float* q  = (const float*)d_in[0];
  const float* k  = (const float*)d_in[1];
  const float* v  = (const float*)d_in[2];
  const float* qs = (const float*)d_in[3];
  const float* ks = (const float*)d_in[4];
  float* out = (float*)d_out;

  const size_t elems = (size_t)B_ * H_ * N_ * D_;
  const size_t need  = elems * sizeof(__bf16) * 3;     // qb + kb + vt = 24 MB

  dim3 agrid(N_ / BM, H_, B_);
  if (ws_size >= need) {
    __bf16* qb = (__bf16*)d_ws;
    __bf16* kb = qb + elems;
    __bf16* vt = kb + elems;
    prep_qk_kernel<<<(B_ * H_ * N_) / 4, 128, 0, stream>>>(q, k, qs, ks, qb, kb);
    prep_v_kernel<<<dim3(N_ / BN, B_ * H_), 128, 0, stream>>>(v, vt);
    flash_bf16_cdna5<<<agrid, 128, 0, stream>>>(qb, kb, vt, out);
  } else {
    flash_fused_cdna5<<<agrid, 128, 0, stream>>>(q, k, v, qs, ks, out);
  }
}
